// MPNNLayer_44641890075286
// MI455X (gfx1250) — compile-verified, matrix operand order checked by reading, not run on hardware
//
#include <hip/hip_runtime.h>

#define DEVI __device__ __forceinline__

typedef __attribute__((ext_vector_type(16))) __bf16 v16bf;
typedef __attribute__((ext_vector_type(8)))  __bf16 v8bf;
typedef __attribute__((ext_vector_type(4)))  __bf16 v4bf;
typedef __attribute__((ext_vector_type(2)))  __bf16 v2bf;
typedef __attribute__((ext_vector_type(8)))  float  v8f;
typedef __attribute__((ext_vector_type(4)))  float  v4f;
typedef __attribute__((ext_vector_type(2)))  float  v2f;

// ---------- bf16 helpers (native casts; backend picks best lowering) ----------
DEVI __bf16 f2bf(float f) { return (__bf16)f; }
DEVI v2bf pk2(float x, float y) {
  v2f t = {x, y};
  return __builtin_convertvector(t, v2bf);
}
DEVI v4bf cvt4(float4 q) {
  v4f t = {q.x, q.y, q.z, q.w};
  return __builtin_convertvector(t, v4bf);
}
DEVI float bf2f(__bf16 h) {
  unsigned short s = __builtin_bit_cast(unsigned short, h);
  unsigned u = ((unsigned)s) << 16;
  return __builtin_bit_cast(float, u);
}
DEVI v8f v8zero() {
  v8f z;
#pragma unroll
  for (int q = 0; q < 8; q++) z[q] = 0.f;
  return z;
}
DEVI v8bf v8bfzero() {
  v8bf z;
#pragma unroll
  for (int q = 0; q < 8; q++) z[q] = f2bf(0.f);
  return z;
}
DEVI v16bf cat8(v8bf lo, v8bf hi) {
  union { v16bf v; v8bf h[2]; } u;
  u.h[0] = lo; u.h[1] = hi;
  return u.v;
}
DEVI v8f wmma_bf(v16bf a, v16bf b, v8f c) {
  return __builtin_amdgcn_wmma_f32_16x16x32_bf16(false, a, false, b, (short)0, c,
                                                 false, false);
}
// 16-bit A/B fragment K index (ISA 7.12.2)
DEVI int fragK(int lane, int e) {
  return ((lane >> 4) << 3) + (e & 7) + ((e >> 3) << 4);
}
// A-fragment from two contiguous 8-float runs (4x float4), packed-converted
DEVI v16bf packA(float4 q0, float4 q1, float4 q2, float4 q3) {
  union { v16bf v; v4bf p[4]; } u;
  u.p[0] = cvt4(q0); u.p[1] = cvt4(q1);
  u.p[2] = cvt4(q2); u.p[3] = cvt4(q3);
  return u.v;
}
// vectorized LayerNorm of a 128-wide LDS row; writes bf16 shadow, optionally f32
DEVI void ln_row(float* row, __bf16* rowb, const float* g, const float* bta,
                 bool write_f32) {
  float m = 0.f;
  for (int c = 0; c < 128; c += 4) {
    float4 x = *(const float4*)(row + c);
    m += x.x + x.y + x.z + x.w;
  }
  m *= (1.f / 128.f);
  float var = 0.f;
  for (int c = 0; c < 128; c += 4) {
    float4 x = *(const float4*)(row + c);
    float d0 = x.x - m, d1 = x.y - m, d2 = x.z - m, d3 = x.w - m;
    var += d0 * d0 + d1 * d1 + d2 * d2 + d3 * d3;
  }
  var *= (1.f / 128.f);
  float inv = rsqrtf(var + 1e-5f);
  unsigned* rb = (unsigned*)rowb;
  for (int c = 0; c < 128; c += 4) {
    float4 x = *(const float4*)(row + c);
    float4 g4 = *(const float4*)(g + c);
    float4 b4 = *(const float4*)(bta + c);
    float4 y;
    y.x = (x.x - m) * inv * g4.x + b4.x;
    y.y = (x.y - m) * inv * g4.y + b4.y;
    y.z = (x.z - m) * inv * g4.z + b4.z;
    y.w = (x.w - m) * inv * g4.w + b4.w;
    if (write_f32) *(float4*)(row + c) = y;
    rb[(c >> 1)]     = __builtin_bit_cast(unsigned, pk2(y.x, y.y));
    rb[(c >> 1) + 1] = __builtin_bit_cast(unsigned, pk2(y.z, y.w));
  }
}

// ---------- weight conversion + B-fragment swizzle ----------
__global__ void k_swz(const float* __restrict__ src, __bf16* __restrict__ dst,
                      int Kp, int Np, int Ksrc, int Nsrc) {
  int idx = blockIdx.x * 256 + threadIdx.x;
  if (idx >= Kp * Np) return;
  int e = idx & 15;
  int lane = (idx >> 4) & 31;
  int t = idx >> 9;
  int nks = Kp >> 5;
  int ks = t % nks;
  int nt = t / nks;
  int k = ks * 32 + fragK(lane, e);
  int c = nt * 16 + (lane & 15);
  float v = (k < Ksrc && c < Nsrc) ? src[k * Nsrc + c] : 0.f;
  dst[idx] = f2bf(v);
}
__global__ void k_cvt(const float* __restrict__ src, __bf16* __restrict__ dst, int n) {
  int i = blockIdx.x * 256 + threadIdx.x;
  if (i < n) dst[i] = f2bf(src[i]);
}

// ---------- msg_g = graph_fts @ W_mg + b_mg  (tiny) ----------
__global__ void k_graph(const float* __restrict__ graph, const float* __restrict__ Wmg,
                        const float* __restrict__ bmg, float* __restrict__ msgg) {
  int t = threadIdx.x;
  int b = t >> 7, c = t & 127;
  float acc = bmg[c];
  for (int k = 0; k < 128; k++) acc += graph[b * 128 + k] * Wmg[k * 128 + c];
  msgg[t] = acc;
}

// ---------- generic WMMA GEMM: Y[M,128] = X[M,128] @ W[128,128] + bias ----------
__global__ void k_gemm128(const float* __restrict__ X, const __bf16* __restrict__ Ws,
                          const float* __restrict__ bias, float* __restrict__ Yf,
                          __bf16* __restrict__ Ybf) {
  __shared__ __align__(32) __bf16 st[8][16][128];   // bf16 output staging
  int wave = threadIdx.x >> 5;
  int lane = threadIdx.x & 31;
  int tile = blockIdx.x * (blockDim.x >> 5) + wave;
  int grp = lane >> 4, col = lane & 15, rowL = lane & 15;
  const float* xr = X + ((long)tile * 16 + rowL) * 128;
  __builtin_prefetch(xr + (long)16 * 128, 0, 1);   // next tile, same lane row
  v16bf a[4];
#pragma unroll
  for (int ks = 0; ks < 4; ks++) {
    int off = ks * 32 + grp * 8;
    a[ks] = packA(*(const float4*)(xr + off), *(const float4*)(xr + off + 4),
                  *(const float4*)(xr + off + 16), *(const float4*)(xr + off + 20));
  }
  v8f acc[8];
#pragma unroll
  for (int nt = 0; nt < 8; nt++) acc[nt] = v8zero();
#pragma unroll
  for (int nt = 0; nt < 8; nt++) {
#pragma unroll
    for (int ks = 0; ks < 4; ks++) {
      v16bf bfr = *(const v16bf*)(Ws + (((nt * 4 + ks) * 32 + lane) << 4));
      acc[nt] = wmma_bf(a[ks], bfr, acc[nt]);
    }
  }
  long base = (long)tile * 16 * 128;
  if (Yf) {
#pragma unroll
    for (int nt = 0; nt < 8; nt++) {
      int c = nt * 16 + col;
      float bb = bias[c];
#pragma unroll
      for (int v = 0; v < 8; v++) Yf[base + (v + grp * 8) * 128 + c] = acc[nt][v] + bb;
    }
  }
  if (Ybf) {
#pragma unroll
    for (int nt = 0; nt < 8; nt++) {
      int c = nt * 16 + col;
      float bb = bias[c];
#pragma unroll
      for (int v = 0; v < 8; v++) st[wave][v + grp * 8][c] = f2bf(acc[nt][v] + bb);
    }
    __syncthreads();
    int cb = grp * 64;
    __bf16* drow = Ybf + base + rowL * 128 + cb;
    const __bf16* srow = &st[wave][rowL][cb];
#pragma unroll
    for (int q = 0; q < 8; q++) *(v8bf*)(drow + q * 8) = *(const v8bf*)(srow + q * 8);
  }
}

// ---------- masked mean reduction over neighbor axis ----------
__global__ void k_reduce(const float* __restrict__ adj, const __bf16* __restrict__ msg_e_bf,
                         const float* __restrict__ msg1, const float* __restrict__ msg2,
                         const float* __restrict__ msgg, float* __restrict__ msgs) {
  __shared__ float sred[128];
  int bn = blockIdx.x;                 // b*256 + n
  int b = bn >> 8, n = bn & 255;
  int c = threadIdx.x;                 // 0..127
  const float* adjb = adj + (long)b * 256 * 256;
  float acc = 0.f, s1 = 0.f;
  for (int m = 0; m < 256; m++) {
    float av = adjb[m * 256 + n];
    s1 += av;
    long ei = ((long)((b << 8) + m) * 256 + n) * 128 + c;
    acc += av * (msg2[((b << 8) + m) * 128 + c] + bf2f(msg_e_bf[ei]));
  }
  sred[c] = adjb[n * 256 + c] + adjb[n * 256 + c + 128];
  __syncthreads();
  for (int s = 64; s > 0; s >>= 1) {
    if (c < s) sred[c] += sred[c + s];
    __syncthreads();
  }
  float D = sred[0];
  float m1 = msg1[bn * 128 + c] + msgg[(b << 7) + c];
  msgs[bn * 128 + c] = (acc + s1 * m1) / D;
}

// ---------- ret = LN(relu(node@W_o1 + msgs@W_o2 + biases)) ----------
__global__ void k_ret(const float* __restrict__ msgs, const __bf16* __restrict__ Ws,
                      const float* __restrict__ pre, const float* __restrict__ bias,
                      const float* __restrict__ lng, const float* __restrict__ lnb,
                      float* __restrict__ Y, __bf16* __restrict__ Ybf) {
  __shared__ __align__(32) float  srow[16][128];
  __shared__ __align__(32) __bf16 srowb[16][128];
  int lane = threadIdx.x & 31;
  int tile = blockIdx.x;
  int grp = lane >> 4, col = lane & 15, rowL = lane & 15;
  const float* xr = msgs + (tile * 16 + rowL) * 128;
  v16bf a[4];
#pragma unroll
  for (int ks = 0; ks < 4; ks++) {
    int off = ks * 32 + grp * 8;
    a[ks] = packA(*(const float4*)(xr + off), *(const float4*)(xr + off + 4),
                  *(const float4*)(xr + off + 16), *(const float4*)(xr + off + 20));
  }
#pragma unroll
  for (int nt = 0; nt < 8; nt++) {
    v8f acc = v8zero();
#pragma unroll
    for (int ks = 0; ks < 4; ks++) {
      v16bf bfr = *(const v16bf*)(Ws + (((nt * 4 + ks) * 32 + lane) << 4));
      acc = wmma_bf(a[ks], bfr, acc);
    }
    int c = nt * 16 + col;
    float bb = bias[c];
#pragma unroll
    for (int v = 0; v < 8; v++) {
      int r = v + grp * 8;
      float y = acc[v] + pre[tile * 2048 + r * 128 + c] + bb;
      srow[r][c] = y > 0.f ? y : 0.f;
    }
  }
  __syncthreads();
  if (lane < 16) ln_row(srow[lane], srowb[lane], lng, lnb, true);
  __syncthreads();
  {
    int cb = grp * 64;
    float* yrow = Y + tile * 2048 + rowL * 128 + cb;
    __bf16* brow = Ybf + tile * 2048 + rowL * 128 + cb;
#pragma unroll
    for (int q = 0; q < 16; q++)
      *(float4*)(yrow + q * 4) = *(const float4*)(&srow[rowL][cb + q * 4]);
#pragma unroll
    for (int q = 0; q < 8; q++)
      *(v8bf*)(brow + q * 8) = *(const v8bf*)(&srowb[rowL][cb + q * 8]);
  }
}

// ---------- fused edge-update pipeline, 16 edges per wave ----------
__global__ void k_edge(const __bf16* __restrict__ msg_e_bf,
                       const __bf16* __restrict__ ret_bf,
                       const __bf16* __restrict__ graph_bf,
                       const __bf16* __restrict__ We1s, const float* __restrict__ be1,
                       const __bf16* __restrict__ We2s, const float* __restrict__ be2,
                       const float* __restrict__ ln1g, const float* __restrict__ ln1b,
                       const __bf16* __restrict__ We3s, const float* __restrict__ be3,
                       const __bf16* __restrict__ We4s, const float* __restrict__ be4,
                       const float* __restrict__ ln2g, const float* __restrict__ ln2b,
                       const __bf16* __restrict__ Wo3s, const float* __restrict__ bo3,
                       float* __restrict__ out_e) {
  __shared__ __align__(32) float  s_rows[4][16][128];
  __shared__ __align__(32) __bf16 s_rows_bf[4][16][128];
  __shared__ __align__(32) __bf16 s_h_bf[4][16][16];
  __shared__ __align__(32) __bf16 s_tg[2][128];    // block-uniform: tgt=ret[b,i], glob
  int w = threadIdx.x >> 5;
  int lane = threadIdx.x & 31;
  int rowL = lane & 15, grp = lane >> 4, col = lane & 15;
  int e0 = blockIdx.x * 64 + w * 16;
  int b = e0 >> 16;
  int i = (e0 >> 8) & 255;
  int j0 = e0 & 255;
  int j = j0 + rowL;
  const __bf16* me_row = msg_e_bf + (size_t)(((b << 8) + i) << 8) * 128;

  // async-stage the two block-uniform concat segments into LDS (ASYNCcnt path)
  if (w == 0) {
    const __bf16* gsrc = (lane < 16)
        ? (ret_bf + (((b << 8) + i) << 7) + rowL * 8)     // tgt = ret[b,i]
        : (graph_bf + (b << 7) + rowL * 8);               // glob = graph[b]
    unsigned ldst = (unsigned)(uintptr_t)&s_tg[lane >> 4][rowL * 8];
    unsigned long long ga = (unsigned long long)(uintptr_t)gsrc;
    asm volatile("global_load_async_to_lds_b128 %0, %1, off"
                 :: "v"(ldst), "v"(ga) : "memory");
  }
  asm volatile("s_wait_asynccnt 0x0" ::: "memory");

  const __bf16* segp[3];
  segp[0] = me_row + ((size_t)j << 7);                                   // msg_e[b,i,j]
  segp[1] = msg_e_bf + ((size_t)((((b << 8) + j) << 8) + i) << 7);       // msg_e[b,j,i]
  segp[2] = ret_bf + (((b << 8) + j) << 7);                              // src = ret[b,j]
  __syncthreads();

  // ---- GEMM1: cat[16,640] @ W_e1[640,16] ----
  v8f acc1 = v8zero();
#pragma unroll
  for (int ks = 0; ks < 20; ks++) {
    int off = (ks & 3) * 32 + grp * 8;
    v8bf r0, r1;
    if (ks < 12) {            // per-lane segments (global, bf16)
      const __bf16* sp = segp[ks >> 2];
      r0 = *(const v8bf*)(sp + off);
      r1 = *(const v8bf*)(sp + off + 16);
    } else {                  // block-uniform segments staged in LDS
      const __bf16* sp = s_tg[(ks >= 16) ? 1 : 0];
      r0 = *(const v8bf*)(sp + off);
      r1 = *(const v8bf*)(sp + off + 16);
    }
    v16bf bfr = *(const v16bf*)(We1s + ((ks * 32 + lane) << 4));
    acc1 = wmma_bf(cat8(r0, r1), bfr, acc1);
  }
  {
    float bb = be1[col];
#pragma unroll
    for (int v = 0; v < 8; v++) {
      float h = acc1[v] + bb;
      s_h_bf[w][v + grp * 8][col] = f2bf(h > 0.f ? h : 0.f);
    }
  }
  __syncthreads();
  // ---- GEMM2: h1[16,16 pad->32] @ W_e2p[32,128] ----
  v16bf a2 = cat8(*(const v8bf*)(&s_h_bf[w][rowL][grp * 8]), v8bfzero());
  v8f acc2[8];
#pragma unroll
  for (int nt = 0; nt < 8; nt++) {
    v16bf bfr = *(const v16bf*)(We2s + ((nt * 32 + lane) << 4));
    acc2[nt] = wmma_bf(a2, bfr, v8zero());
  }
#pragma unroll
  for (int nt = 0; nt < 8; nt++) {
    int c = nt * 16 + col;
    float bb = be2[c];
#pragma unroll
    for (int v = 0; v < 8; v++) s_rows[w][v + grp * 8][c] = acc2[nt][v] + bb;
  }
  __syncthreads();
  {
    int cb = grp * 64;
    const __bf16* mrow = me_row + ((size_t)(j0 + rowL) << 7) + cb;
    float* srow = &s_rows[w][rowL][cb];
#pragma unroll
    for (int q = 0; q < 8; q++) {
      v8bf mv = *(const v8bf*)(mrow + q * 8);
#pragma unroll
      for (int t = 0; t < 8; t++) srow[q * 8 + t] += bf2f(mv[t]);
    }
  }
  __syncthreads();
  if (lane < 16) ln_row(s_rows[w][lane], s_rows_bf[w][lane], ln1g, ln1b, true);
  __syncthreads();
  // ---- GEMM3: msg_e2[16,128] @ W_e3p[128,16] ----
  v8f acc3 = v8zero();
#pragma unroll
  for (int ks = 0; ks < 4; ks++) {
    int off = ks * 32 + grp * 8;
    v8bf r0 = *(const v8bf*)(&s_rows_bf[w][rowL][off]);
    v8bf r1 = *(const v8bf*)(&s_rows_bf[w][rowL][off + 16]);
    v16bf bfr = *(const v16bf*)(We3s + ((ks * 32 + lane) << 4));
    acc3 = wmma_bf(cat8(r0, r1), bfr, acc3);
  }
  __syncthreads();
  {
    float bb = (col < 8) ? be3[col] : 0.f;
#pragma unroll
    for (int v = 0; v < 8; v++) {
      float h = acc3[v] + bb;
      s_h_bf[w][v + grp * 8][col] = f2bf(h > 0.f ? h : 0.f);
    }
  }
  __syncthreads();
  // ---- GEMM4: h2[16,16 pad->32] @ W_e4p[32,128] ----
  v16bf a4 = cat8(*(const v8bf*)(&s_h_bf[w][rowL][grp * 8]), v8bfzero());
  v8f acc4[8];
#pragma unroll
  for (int nt = 0; nt < 8; nt++) {
    v16bf bfr = *(const v16bf*)(We4s + ((nt * 32 + lane) << 4));
    acc4[nt] = wmma_bf(a4, bfr, v8zero());
  }
  __syncthreads();
#pragma unroll
  for (int nt = 0; nt < 8; nt++) {
    int c = nt * 16 + col;
    float bb = be4[c];
#pragma unroll
    for (int v = 0; v < 8; v++) s_rows[w][v + grp * 8][c] += acc4[nt][v] + bb;
  }
  __syncthreads();
  if (lane < 16) ln_row(s_rows[w][lane], s_rows_bf[w][lane], ln2g, ln2b, false);
  __syncthreads();
  // ---- GEMM5: m3[16,128] @ W_o3[128,128] + b_o3 ----
  v16bf a5[4];
#pragma unroll
  for (int ks = 0; ks < 4; ks++) {
    int off = ks * 32 + grp * 8;
    a5[ks] = cat8(*(const v8bf*)(&s_rows_bf[w][rowL][off]),
                  *(const v8bf*)(&s_rows_bf[w][rowL][off + 16]));
  }
#pragma unroll
  for (int nt = 0; nt < 8; nt++) {
    v8f acc5 = v8zero();
#pragma unroll
    for (int ks = 0; ks < 4; ks++) {
      v16bf bfr = *(const v16bf*)(Wo3s + (((nt * 4 + ks) * 32 + lane) << 4));
      acc5 = wmma_bf(a5[ks], bfr, acc5);
    }
    int c = nt * 16 + col;
    float bb = bo3[c];
#pragma unroll
    for (int v = 0; v < 8; v++) s_rows[w][v + grp * 8][c] = acc5[v] + bb;
  }
  __syncthreads();
  {
    int cb = grp * 64;
    const float* srow = &s_rows[w][rowL][cb];
    float* orow = out_e + ((size_t)(e0 + rowL) << 7) + cb;
#pragma unroll
    for (int q = 0; q < 16; q++) *(float4*)(orow + q * 4) = *(const float4*)(srow + q * 4);
  }
}

extern "C" void kernel_launch(void* const* d_in, const int* in_sizes, int n_in,
                              void* d_out, int out_size, void* d_ws, size_t ws_size,
                              hipStream_t stream) {
  (void)in_sizes; (void)n_in; (void)out_size; (void)ws_size;
  const float* node  = (const float*)d_in[0];
  const float* edge  = (const float*)d_in[1];
  const float* graph = (const float*)d_in[2];
  const float* adj   = (const float*)d_in[3];
  const float* W_m1 = (const float*)d_in[6];  const float* b_m1 = (const float*)d_in[7];
  const float* W_m2 = (const float*)d_in[8];  const float* b_m2 = (const float*)d_in[9];
  const float* W_me = (const float*)d_in[10]; const float* b_me = (const float*)d_in[11];
  const float* W_mg = (const float*)d_in[12]; const float* b_mg = (const float*)d_in[13];
  const float* W_o1 = (const float*)d_in[14]; const float* b_o1 = (const float*)d_in[15];
  const float* W_o2 = (const float*)d_in[16]; const float* b_o2 = (const float*)d_in[17];
  const float* W_o3 = (const float*)d_in[18]; const float* b_o3 = (const float*)d_in[19];
  const float* ln_g = (const float*)d_in[20]; const float* ln_b = (const float*)d_in[21];
  const float* W_e1 = (const float*)d_in[22]; const float* b_e1 = (const float*)d_in[23];
  const float* W_e2 = (const float*)d_in[24]; const float* b_e2 = (const float*)d_in[25];
  const float* ln1g = (const float*)d_in[26]; const float* ln1b = (const float*)d_in[27];
  const float* W_e3 = (const float*)d_in[28]; const float* b_e3 = (const float*)d_in[29];
  const float* W_e4 = (const float*)d_in[30]; const float* b_e4 = (const float*)d_in[31];
  const float* ln2g = (const float*)d_in[32]; const float* ln2b = (const float*)d_in[33];

  char* ws = (char*)d_ws;
  size_t off = 0;
  auto take = [&](size_t bytes) {
    void* p = ws + off;
    off = (off + bytes + 255) & ~(size_t)255;
    return p;
  };
  __bf16* msg_e_bf = (__bf16*)take(16777216ull * 2);  // [B,N,N,128] bf16
  float* msg1   = (float*)take(65536 * 4);
  float* msg2   = (float*)take(65536 * 4);
  float* nodeW1 = (float*)take(65536 * 4);
  float* msgs   = (float*)take(65536 * 4);
  float* msgg   = (float*)take(256 * 4);
  __bf16* ret_bf   = (__bf16*)take(65536 * 2);
  __bf16* graph_bf = (__bf16*)take(256 * 2);
  __bf16* sW_m1 = (__bf16*)take(16384 * 2);
  __bf16* sW_m2 = (__bf16*)take(16384 * 2);
  __bf16* sW_me = (__bf16*)take(16384 * 2);
  __bf16* sW_o1 = (__bf16*)take(16384 * 2);
  __bf16* sW_o2 = (__bf16*)take(16384 * 2);
  __bf16* sW_o3 = (__bf16*)take(16384 * 2);
  __bf16* sW_e1 = (__bf16*)take(10240 * 2);
  __bf16* sW_e2 = (__bf16*)take(4096 * 2);
  __bf16* sW_e3 = (__bf16*)take(2048 * 2);
  __bf16* sW_e4 = (__bf16*)take(4096 * 2);

  float* ret_out  = (float*)d_out;        // [B,N,128]
  float* edge_out = ret_out + 65536;      // [B,N,N,128]

  k_swz<<<64, 256, 0, stream>>>(W_m1, sW_m1, 128, 128, 128, 128);
  k_swz<<<64, 256, 0, stream>>>(W_m2, sW_m2, 128, 128, 128, 128);
  k_swz<<<64, 256, 0, stream>>>(W_me, sW_me, 128, 128, 128, 128);
  k_swz<<<64, 256, 0, stream>>>(W_o1, sW_o1, 128, 128, 128, 128);
  k_swz<<<64, 256, 0, stream>>>(W_o2, sW_o2, 128, 128, 128, 128);
  k_swz<<<64, 256, 0, stream>>>(W_o3, sW_o3, 128, 128, 128, 128);
  k_swz<<<40, 256, 0, stream>>>(W_e1, sW_e1, 640, 16, 640, 16);
  k_swz<<<16, 256, 0, stream>>>(W_e2, sW_e2, 32, 128, 16, 128);
  k_swz<<<8, 256, 0, stream>>>(W_e3, sW_e3, 128, 16, 128, 8);
  k_swz<<<16, 256, 0, stream>>>(W_e4, sW_e4, 32, 128, 8, 128);
  k_cvt<<<1, 256, 0, stream>>>(graph, graph_bf, 256);

  k_graph<<<1, 256, 0, stream>>>(graph, W_mg, b_mg, msgg);
  k_gemm128<<<4, 256, 0, stream>>>(node, sW_m1, b_m1, msg1, nullptr);
  k_gemm128<<<4, 256, 0, stream>>>(node, sW_m2, b_m2, msg2, nullptr);
  k_gemm128<<<4, 256, 0, stream>>>(node, sW_o1, b_o1, nodeW1, nullptr);
  k_gemm128<<<1024, 256, 0, stream>>>(edge, sW_me, b_me, nullptr, msg_e_bf);
  k_reduce<<<512, 128, 0, stream>>>(adj, msg_e_bf, msg1, msg2, msgg, msgs);
  k_ret<<<32, 32, 0, stream>>>(msgs, sW_o2, nodeW1, b_o2, ln_g, ln_b, ret_out, ret_bf);
  k_edge<<<2048, 128, 0, stream>>>(msg_e_bf, ret_bf, graph_bf, sW_e1, b_e1, sW_e2, b_e2,
                                   ln1g, ln1b, sW_e3, b_e3, sW_e4, b_e4, ln2g, ln2b,
                                   sW_o3, b_o3, edge_out);
}
